// GCGRU_44976897524060
// MI455X (gfx1250) — matrix operand order repaired and checked by reading, not compiled, not run on hardware
//
#include <hip/hip_runtime.h>
#include <math.h>
#include <stdint.h>

typedef __attribute__((ext_vector_type(16))) _Float16 v16h;
typedef __attribute__((ext_vector_type(8)))  float    v8f;

#define B_   2
#define T_   16
#define N_   10000
#define IN_  128
#define HID_ 256
#define OUT_ 128
#define E_   160000

// ---------------- utility kernels ----------------
__global__ void k_fill(float* __restrict__ p, float v, int n) {
  int i = blockIdx.x * blockDim.x + threadIdx.x;
  if (i < n) p[i] = v;
}

__global__ void k_fill_h16(_Float16* __restrict__ p, int n) {
  int i = blockIdx.x * blockDim.x + threadIdx.x;
  if (i < n) p[i] = (_Float16)0.f;
}

__global__ void k_f32_to_f16(const float* __restrict__ s, _Float16* __restrict__ d, int n) {
  int i = blockIdx.x * blockDim.x + threadIdx.x;
  if (i < n) d[i] = (_Float16)s[i];
}

__global__ void k_deg_accum(const int* __restrict__ dst, float* __restrict__ deg, int e) {
  int i = blockIdx.x * blockDim.x + threadIdx.x;
  if (i < e) atomicAdd(&deg[dst[i]], 1.0f);
}

__global__ void k_rsqrt_inplace(float* __restrict__ d, int n) {
  int i = blockIdx.x * blockDim.x + threadIdx.x;
  if (i < n) d[i] = rsqrtf(fmaxf(d[i], 1e-12f));
}

__global__ void k_edge_norm(const int* __restrict__ ei, const float* __restrict__ dinv,
                            float* __restrict__ norm, int e) {
  int i = blockIdx.x * blockDim.x + threadIdx.x;
  if (i < e) norm[i] = dinv[ei[i]] * dinv[ei[E_ + i]];
}

// ---- transposed f16 weight builders: Bt[col][k] so GEMM B-tiles are row-contiguous ----
__global__ void k_build_wxT(const float* __restrict__ Wr, const float* __restrict__ Wz,
                            const float* __restrict__ Wn, _Float16* __restrict__ WT) {
  int i = blockIdx.x * blockDim.x + threadIdx.x;      // 768 * 128
  if (i >= 768 * IN_) return;
  int c = i / IN_, k = i % IN_;
  const float* src = (c < 256) ? Wr : (c < 512) ? Wz : Wn;
  WT[i] = (_Float16)src[(size_t)k * 256 + (c & 255)];
}

__global__ void k_build_whrzT(const float* __restrict__ Wr, const float* __restrict__ Wz,
                              _Float16* __restrict__ WT) {
  int i = blockIdx.x * blockDim.x + threadIdx.x;      // 512 * 256
  if (i >= 512 * HID_) return;
  int c = i / HID_, k = i % HID_;
  const float* src = (c < 256) ? Wr : Wz;
  WT[i] = (_Float16)src[(size_t)k * 256 + (c & 255)];
}

// W[R][C] -> WT[C][R] (f16)
__global__ void k_transpose_f16(const float* __restrict__ W, _Float16* __restrict__ WT,
                                int R, int Cc) {
  int i = blockIdx.x * blockDim.x + threadIdx.x;
  if (i >= R * Cc) return;
  int c = i / R, r = i % R;
  WT[i] = (_Float16)W[(size_t)r * Cc + c];
}

// ---------------- async LDS staging helper ----------------
// One 16-byte chunk per lane via CDNA5 global_load_async_to_lds_b128 (ASYNCcnt-tracked DMA).
__device__ __forceinline__ void async_chunk16(uint32_t lds_addr, uint64_t gaddr) {
  asm volatile("global_load_async_to_lds_b128 %0, %1, off"
               :: "v"(lds_addr), "v"(gaddr) : "memory");
}

// ---------------- WMMA GEMM ----------------
// C[M, cOfs + gridDim.y*128) (ldC, f32) = A[M,K] (f16, row-major) x Bt[Ncols][K]^T (f16, col-major-of-B)
// Block tile 64x128, 8 waves; double-buffered LDS filled by async-to-LDS DMA; 3 async ops/wave/K-step.
__global__ __launch_bounds__(256)
void k_gemm_wmma(const _Float16* __restrict__ A, const _Float16* __restrict__ Bt,
                 float* __restrict__ C, const float* __restrict__ bias,
                 int M, int K, int ldC, int cOfs, int accFlag)
{
  __shared__ __align__(16) _Float16 As[2][64][40];    // 64 x 32 halves (+8 pad)
  __shared__ __align__(16) _Float16 Bs[2][128][40];   // [n][k], rows of Bt

  const int tid  = threadIdx.x;
  const int m0   = blockIdx.x * 64;
  const int n0   = blockIdx.y * 128;
  const int lane = tid & 31;
  const int w    = tid >> 5;
  const int wm   = w >> 1;           // 0..3
  const int wn   = w & 1;            // 0..1
  const int lm   = lane & 15;
  const int lh   = lane >> 4;

  // per-thread chunk coordinates (constant across K-steps)
  const int ar = tid >> 2, ac = (tid & 3) << 3;                 // A: row, half-offset
  int agrow = m0 + ar; if (agrow > M - 1) agrow = M - 1;        // clamp (no divergence)
  const int br0 = tid >> 2,           bc0 = (tid & 3) << 3;     // B chunk 0
  const int br1 = (tid + 256) >> 2,   bc1 = (tid & 3) << 3;     // B chunk 1

  v8f acc[4] = {};
  const int KT = K >> 5;

  // issue stage for k-step 0 into buffer 0
  {
    async_chunk16((uint32_t)(uintptr_t)(&As[0][ar][ac]),
                  (uint64_t)(uintptr_t)(A + (size_t)agrow * K + ac));
    async_chunk16((uint32_t)(uintptr_t)(&Bs[0][br0][bc0]),
                  (uint64_t)(uintptr_t)(Bt + (size_t)(n0 + br0) * K + bc0));
    async_chunk16((uint32_t)(uintptr_t)(&Bs[0][br1][bc1]),
                  (uint64_t)(uintptr_t)(Bt + (size_t)(n0 + br1) * K + bc1));
  }

  int buf = 0;
  for (int kt = 0; kt < KT; ++kt) {
    if (kt + 1 < KT) {
      const int k1 = (kt + 1) << 5;
      const int nb = buf ^ 1;
      async_chunk16((uint32_t)(uintptr_t)(&As[nb][ar][ac]),
                    (uint64_t)(uintptr_t)(A + (size_t)agrow * K + k1 + ac));
      async_chunk16((uint32_t)(uintptr_t)(&Bs[nb][br0][bc0]),
                    (uint64_t)(uintptr_t)(Bt + (size_t)(n0 + br0) * K + k1 + bc0));
      async_chunk16((uint32_t)(uintptr_t)(&Bs[nb][br1][bc1]),
                    (uint64_t)(uintptr_t)(Bt + (size_t)(n0 + br1) * K + k1 + bc1));
      asm volatile("s_wait_asynccnt 0x3" ::: "memory");  // oldest 3 (current buffer) done
    } else {
      asm volatile("s_wait_asynccnt 0x0" ::: "memory");
    }
    __syncthreads();

    // A fragment: m = lm; lanes<16 K {0..15}, lanes>=16 K {16..31} (contiguous, matches B)
    v16h a;
    const _Float16* arr = &As[buf][wm * 16 + lm][0];
    const int kb = lh * 16;
#pragma unroll
    for (int j = 0; j < 8; ++j) { a[j] = arr[kb + j]; a[8 + j] = arr[kb + 8 + j]; }

#pragma unroll
    for (int nt = 0; nt < 4; ++nt) {
      // B fragment: n = lm; contiguous K run of 16 halves at kb2 in [n][k] LDS layout
      const _Float16* brr = &Bs[buf][wn * 64 + nt * 16 + lm][0];
      const int kb2 = lh * 16;
      v16h bf;
#pragma unroll
      for (int j = 0; j < 16; ++j) bf[j] = brr[kb2 + j];
      acc[nt] = __builtin_amdgcn_wmma_f32_16x16x32_f16(
          false, a, false, bf, (short)0, acc[nt], false, false);
    }
    __syncthreads();   // before DMA may overwrite this buffer next iteration
    buf ^= 1;
  }

  // epilogue: C element r -> m = r + 8*lh, n = lm
#pragma unroll
  for (int nt = 0; nt < 4; ++nt) {
#pragma unroll
    for (int r = 0; r < 8; ++r) {
      int row = m0 + wm * 16 + lh * 8 + r;
      int col = n0 + wn * 64 + nt * 16 + lm;
      if (row < M) {
        size_t o = (size_t)row * ldC + cOfs + col;
        float v = acc[nt][r];
        if (bias) v += bias[col];
        C[o] = accFlag ? (C[o] + v) : v;
      }
    }
  }
}

// ---------------- aggregation ----------------
__global__ void k_agg_init(const float* __restrict__ P, float* __restrict__ Q,
                           const float* __restrict__ dinv) {
  int idx = blockIdx.x * blockDim.x + threadIdx.x;
  if (idx < N_ * 1024) {
    int i = idx >> 10;
    float s = dinv[i];
    Q[idx] = s * s * P[idx];
  }
}

__global__ void k_agg_edges(const float* __restrict__ P, float* __restrict__ Q,
                            const int* __restrict__ ei, const float* __restrict__ norm) {
  int e = blockIdx.x;
  int s = ei[e], d = ei[E_ + e];
  float wgt = norm[e];
  const float* ps = P + (size_t)s * 1024;
  float* qd = Q + (size_t)d * 1024;
  for (int c = threadIdx.x; c < 1024; c += blockDim.x)
    atomicAdd(&qd[c], wgt * ps[c]);
}

// ---------------- GRU gates: h (f16) updated in place ----------------
__global__ void k_gate(const float* __restrict__ Q, _Float16* __restrict__ h,
                       const float* __restrict__ b_xr, const float* __restrict__ b_hr,
                       const float* __restrict__ b_xz, const float* __restrict__ b_hz,
                       const float* __restrict__ b_xn, const float* __restrict__ b_hn) {
  int idx = blockIdx.x * blockDim.x + threadIdx.x;
  if (idx >= N_ * HID_) return;
  int i = idx >> 8, c = idx & 255;
  const float* q = Q + (size_t)i * 1024;
  float r = 1.f / (1.f + __expf(-(q[c]       + b_xr[c] + b_hr[c])));
  float z = 1.f / (1.f + __expf(-(q[256 + c] + b_xz[c] + b_hz[c])));
  float n = tanhf(q[512 + c] + b_xn[c] + r * (q[768 + c] + b_hn[c]));
  float hv = (float)h[idx];
  h[idx] = (_Float16)((1.f - z) * hv + z * n);
}

// ---------------- host orchestration ----------------
extern "C" void kernel_launch(void* const* d_in, const int* in_sizes, int n_in,
                              void* d_out, int out_size, void* d_ws, size_t ws_size,
                              hipStream_t stream) {
  const float* x    = (const float*)d_in[0];
  const int*   ei   = (const int*)  d_in[1];
  const float* W_xr = (const float*)d_in[2];
  const float* b_xr = (const float*)d_in[3];
  const float* W_hr = (const float*)d_in[4];
  const float* b_hr = (const float*)d_in[5];
  const float* W_xz = (const float*)d_in[6];
  const float* b_xz = (const float*)d_in[7];
  const float* W_hz = (const float*)d_in[8];
  const float* b_hz = (const float*)d_in[9];
  const float* W_xn = (const float*)d_in[10];
  const float* b_xn = (const float*)d_in[11];
  const float* W_hn = (const float*)d_in[12];
  const float* b_hn = (const float*)d_in[13];
  const float* W_fc = (const float*)d_in[14];
  const float* b_fc = (const float*)d_in[15];
  float* out = (float*)d_out;

  char* ws = (char*)d_ws;
  size_t off = 0;
  auto carve = [&](size_t bytes) -> void* {
    void* p = ws + off;
    off += (bytes + 255) & ~(size_t)255;
    return p;
  };
  float*     dinv   = (float*)    carve((size_t)N_ * 4);
  float*     norm   = (float*)    carve((size_t)E_ * 4);
  _Float16*  WxT    = (_Float16*) carve((size_t)768 * IN_  * 2);   // [768][128]
  _Float16*  WhrzT  = (_Float16*) carve((size_t)512 * HID_ * 2);   // [512][256]
  _Float16*  WhnT   = (_Float16*) carve((size_t)256 * HID_ * 2);   // [256][256]
  _Float16*  WfcT   = (_Float16*) carve((size_t)OUT_ * HID_ * 2);  // [128][256]
  _Float16*  xh     = (_Float16*) carve((size_t)B_ * T_ * N_ * IN_ * 2);
  _Float16*  h16    = (_Float16*) carve((size_t)B_ * N_ * HID_ * 2);
  float*     P      = (float*)    carve((size_t)N_ * 1024 * 4);
  float*     Q      = (float*)    carve((size_t)N_ * 1024 * 4);

  // ---- setup (recomputed each call: deterministic, capture-safe) ----
  k_fill<<<(N_ + 255) / 256, 256, 0, stream>>>(dinv, 1.0f, N_);
  k_deg_accum<<<(E_ + 255) / 256, 256, 0, stream>>>(ei + E_, dinv, E_);
  k_rsqrt_inplace<<<(N_ + 255) / 256, 256, 0, stream>>>(dinv, N_);
  k_edge_norm<<<(E_ + 255) / 256, 256, 0, stream>>>(ei, dinv, norm, E_);
  k_build_wxT<<<(768 * IN_ + 255) / 256, 256, 0, stream>>>(W_xr, W_xz, W_xn, WxT);
  k_build_whrzT<<<(512 * HID_ + 255) / 256, 256, 0, stream>>>(W_hr, W_hz, WhrzT);
  k_transpose_f16<<<(HID_ * HID_ + 255) / 256, 256, 0, stream>>>(W_hn, WhnT, HID_, HID_);
  k_transpose_f16<<<(HID_ * OUT_ + 255) / 256, 256, 0, stream>>>(W_fc, WfcT, HID_, OUT_);
  {
    int nx = B_ * T_ * N_ * IN_;
    k_f32_to_f16<<<(nx + 255) / 256, 256, 0, stream>>>(x, xh, nx);
  }
  k_fill_h16<<<(B_ * N_ * HID_ + 255) / 256, 256, 0, stream>>>(h16, B_ * N_ * HID_);

  const int MB = (N_ + 63) / 64;
  dim3 blk(256);

  for (int t = 0; t < T_; ++t) {
    for (int b = 0; b < B_; ++b) {
      const _Float16* xt = xh + ((size_t)(b * T_ + t)) * N_ * IN_;
      _Float16* hb = h16 + (size_t)b * N_ * HID_;

      // P[:,0:768)    = xt @ [W_xr|W_xz|W_xn]
      k_gemm_wmma<<<dim3(MB, 6), blk, 0, stream>>>(
          xt, WxT, P, nullptr, N_, IN_, 1024, 0, 0);
      // P[:,0:512)   += h  @ [W_hr|W_hz]
      k_gemm_wmma<<<dim3(MB, 4), blk, 0, stream>>>(
          hb, WhrzT, P, nullptr, N_, HID_, 1024, 0, 1);
      // P[:,768:1024) = h  @ W_hn
      k_gemm_wmma<<<dim3(MB, 2), blk, 0, stream>>>(
          hb, WhnT, P, nullptr, N_, HID_, 1024, 768, 0);

      // Q = D^-1/2 (A+I) D^-1/2 over all 1024 fused channels (P+Q L2-resident)
      k_agg_init<<<(N_ * 1024 + 255) / 256, 256, 0, stream>>>(P, Q, dinv);
      k_agg_edges<<<E_, 256, 0, stream>>>(P, Q, ei, norm);

      // gates -> h (f16, in place)
      k_gate<<<(N_ * HID_ + 255) / 256, 256, 0, stream>>>(
          Q, hb, b_xr, b_hr, b_xz, b_hz, b_xn, b_hn);

      // out[b,t] = h @ W_fc + b_fc
      float* ot = out + ((size_t)(b * T_ + t)) * N_ * OUT_;
      k_gemm_wmma<<<dim3(MB, 1), blk, 0, stream>>>(
          hb, WfcT, ot, b_fc, N_, HID_, OUT_, 0, 0);
    }
  }
}
